// Causal2X_9869834846379
// MI455X (gfx1250) — matrix operand order repaired and checked by reading, not compiled, not run on hardware
//
#include <hip/hip_runtime.h>
#include <math.h>

// ---------------- problem constants (from reference) ----------------
constexpr int kN = 102400;   // nodes
constexpr int kE = 500000;   // edges (divisible by 32)
constexpr int kH = 128;      // hidden
constexpr int kB = 512;      // graphs
constexpr int kL = 10;       // labels

// ---------------- WMMA types ----------------
typedef __attribute__((ext_vector_type(16))) __bf16 v16bf;
typedef __attribute__((ext_vector_type(8)))  float  v8f;

union Frag {
    v16bf v;
    uint4 q[2];
};

// ---------------- workspace layout (bytes) ----------------
constexpr size_t OFF_W1T    = 0;                            // [512][256] bf16
constexpr size_t OFF_W2T    = OFF_W1T + 512 * 256 * 2;      // [256][512] bf16
constexpr size_t OFF_W3T    = OFF_W2T + 256 * 512 * 2;      // [128][256] bf16
constexpr size_t OFF_W4T    = OFF_W3T + 128 * 256 * 2;      // [128][128] bf16
constexpr size_t OFF_W5T    = OFF_W4T + 128 * 128 * 2;      // [10][128]  bf16
constexpr size_t OFF_SCORES = OFF_W5T + ((10 * 128 * 2 + 15) & ~size_t(15)); // E f32
constexpr size_t OFF_EBATCH = OFF_SCORES + size_t(kE) * 4;  // E i32
constexpr size_t OFF_PM     = OFF_EBATCH + size_t(kE) * 4;  // 1024 f32
constexpr size_t OFF_PS     = OFF_PM + 1024 * 4;            // 1024 f32
constexpr size_t OFF_MS     = OFF_PS + 1024 * 4;            // 2 f32 (padded 16)
constexpr size_t OFF_SEGU   = OFF_MS + 16;                  // B u32
constexpr size_t OFF_SEGA   = OFF_SEGU + size_t(kB) * 4;    // B i32

// ---------------- helpers ----------------
__device__ __forceinline__ unsigned short f2bf(float f) {
    unsigned u = __float_as_uint(f);
    unsigned r = u + 0x7FFFu + ((u >> 16) & 1u);   // round-to-nearest-even
    return (unsigned short)(r >> 16);
}

__device__ __forceinline__ float elu(float x) {
    // branchless: max(x,0) + min(expm1(x),0)
    return fmaxf(x, 0.f) + fminf(expm1f(x), 0.f);
}

__device__ __forceinline__ unsigned enc_order(float s) {
    unsigned b = __float_as_uint(s);
    return (b & 0x80000000u) ? ~b : (b | 0x80000000u);
}
__device__ __forceinline__ float dec_order(unsigned u) {
    return (u & 0x80000000u) ? __uint_as_float(u & 0x7FFFFFFFu)
                             : __uint_as_float(~u);
}

// ---------------- kernel 1: weight transpose + bf16 quantize ----------------
__global__ void prep_weights(const float* __restrict__ W1, const float* __restrict__ W2,
                             const float* __restrict__ W3, const float* __restrict__ W4,
                             const float* __restrict__ W5,
                             unsigned short* __restrict__ W1T, unsigned short* __restrict__ W2T,
                             unsigned short* __restrict__ W3T, unsigned short* __restrict__ W4T,
                             unsigned short* __restrict__ W5T) {
    int idx = blockIdx.x * blockDim.x + threadIdx.x;
    const int S1 = 512 * 256, S2 = 256 * 512, S3 = 128 * 256, S4 = 128 * 128, S5 = 10 * 128;
    if (idx < S1) {
        int n = idx / 256, k = idx % 256;
        W1T[idx] = f2bf(W1[k * 512 + n]);
    } else if (idx < S1 + S2) {
        int j = idx - S1; int n = j / 512, k = j % 512;
        W2T[j] = f2bf(W2[k * 256 + n]);
    } else if (idx < S1 + S2 + S3) {
        int j = idx - S1 - S2; int n = j / 256, k = j % 256;
        W3T[j] = f2bf(W3[k * 128 + n]);
    } else if (idx < S1 + S2 + S3 + S4) {
        int j = idx - S1 - S2 - S3; int n = j / 128, k = j % 128;
        W4T[j] = f2bf(W4[k * 128 + n]);
    } else if (idx < S1 + S2 + S3 + S4 + S5) {
        int j = idx - S1 - S2 - S3 - S4; int n = j / 128, k = j % 128;
        W5T[j] = f2bf(W5[k * 10 + n]);
    }
}

// ------- WMMA GEMM layer: [32,K] (LDS bf16) x [K,N] (Wt global [N][K] bf16) -------
// Two M-tiles (32 edges) share each B fragment: 2 WMMAs per 32B/lane of weight load.
template<int N, int K, bool ELU>
__device__ __forceinline__ void gemm_layer(const unsigned short* sIn,            // LDS [32][K]
                                           const unsigned short* __restrict__ Wt,// global [N][K]
                                           const float* __restrict__ bias,       // global [N]
                                           unsigned short* sOut,                 // LDS [32][N]
                                           int wave, int lane) {
    const int half = lane >> 4;
    const int nl   = lane & 15;
    const unsigned short* aRow0 = sIn + nl * K;          // tile 0: rows 0..15
    const unsigned short* aRow1 = sIn + (16 + nl) * K;   // tile 1: rows 16..31
    for (int n0 = wave * 16; n0 < N; n0 += 128) {        // 8 waves cover N
        float bn = bias[n0 + nl];
        v8f acc0 = {bn, bn, bn, bn, bn, bn, bn, bn};
        v8f acc1 = acc0;
        const unsigned short* bRow = Wt + (size_t)(n0 + nl) * K + 8 * half;
#pragma unroll
        for (int k0 = 0; k0 < K; k0 += 32) {
            Frag b;
            b.q[0] = *(const uint4*)(bRow + k0);
            b.q[1] = *(const uint4*)(bRow + k0 + 16);
            Frag a0, a1;
            const unsigned short* ap0 = aRow0 + k0 + 8 * half;
            a0.q[0] = *(const uint4*)(ap0);
            a0.q[1] = *(const uint4*)(ap0 + 16);
            const unsigned short* ap1 = aRow1 + k0 + 8 * half;
            a1.q[0] = *(const uint4*)(ap1);
            a1.q[1] = *(const uint4*)(ap1 + 16);
            acc0 = __builtin_amdgcn_wmma_f32_16x16x32_bf16(false, a0.v, false, b.v,
                                                           (short)0, acc0, false, false);
            acc1 = __builtin_amdgcn_wmma_f32_16x16x32_bf16(false, a1.v, false, b.v,
                                                           (short)0, acc1, false, false);
        }
#pragma unroll
        for (int r = 0; r < 8; ++r) {
            float v0 = acc0[r], v1 = acc1[r];
            if (ELU) { v0 = elu(v0); v1 = elu(v1); }
            int m = r + 8 * half;
            sOut[m * N + n0 + nl]        = f2bf(v0);
            sOut[(16 + m) * N + n0 + nl] = f2bf(v1);
        }
    }
}

// ---------------- kernel 2: per-edge MLP chain (32 edges / block, 8 waves) ----------------
__global__ __launch_bounds__(256)
void edge_mlp(const float* __restrict__ node_reps, const int* __restrict__ edge_index,
              const int* __restrict__ batch, const int* __restrict__ y,
              const float* __restrict__ b1, const float* __restrict__ b2,
              const float* __restrict__ b3, const float* __restrict__ b4,
              const float* __restrict__ b5,
              const unsigned short* __restrict__ W1T, const unsigned short* __restrict__ W2T,
              const unsigned short* __restrict__ W3T, const unsigned short* __restrict__ W4T,
              const unsigned short* __restrict__ W5T,
              float* __restrict__ scores, int* __restrict__ ebatch) {
    __shared__ unsigned short sA0[32 * 256];
    __shared__ unsigned short sH1[32 * 512];
    __shared__ unsigned short sH2[32 * 256];
    __shared__ unsigned short sAct[32 * 128];
    __shared__ unsigned short sH4[32 * 128];
    __shared__ float sD[32 * 16];
    __shared__ int sSrc[32], sDst[32], sG[32], sY[32];

    const int tid  = threadIdx.x;
    const int wave = tid >> 5;
    const int lane = tid & 31;
    const int base = blockIdx.x * 32;

    if (tid < 32) {
        int e = base + tid;
        int s = edge_index[e];
        int d = edge_index[kE + e];
        sSrc[tid] = s;
        sDst[tid] = d;
        int g = batch[s];
        sG[tid] = g;
        sY[tid] = y[g];
    }
    __syncthreads();

    // gather + concat + bf16 quantize: [32][256]
    for (int idx = tid; idx < 32 * 256; idx += 256) {
        int e = idx >> 8;
        int f = idx & 255;
        int node = (f < 128) ? sSrc[e] : sDst[e];
        sA0[idx] = f2bf(node_reps[(size_t)node * kH + (f & 127)]);
    }
    __syncthreads();

    gemm_layer<512, 256, true >(sA0, W1T, b1, sH1, wave, lane);  __syncthreads();
    gemm_layer<256, 512, true >(sH1, W2T, b2, sH2, wave, lane);  __syncthreads();
    gemm_layer<128, 256, false>(sH2, W3T, b3, sAct, wave, lane); __syncthreads();
    gemm_layer<128, 128, true >(sAct, W4T, b4, sH4, wave, lane); __syncthreads();

    // layer 5 collapsed to per-edge label column: diagonal 16x16 WMMA tiles (waves 0,1)
    if (wave < 2) {
        const int half = lane >> 4;
        const int nl   = lane & 15;
        v8f acc = {0.f, 0.f, 0.f, 0.f, 0.f, 0.f, 0.f, 0.f};
        const unsigned short* aRow = sH4 + (16 * wave + nl) * kH;
        const unsigned short* bRow = W5T + (size_t)sY[16 * wave + nl] * kH + 8 * half;
#pragma unroll
        for (int k0 = 0; k0 < kH; k0 += 32) {
            Frag a, b;
            const unsigned short* ap = aRow + k0 + 8 * half;
            a.q[0] = *(const uint4*)(ap);
            a.q[1] = *(const uint4*)(ap + 16);
            const unsigned short* bp = bRow + k0;
            b.q[0] = *(const uint4*)(bp);
            b.q[1] = *(const uint4*)(bp + 16);
            acc = __builtin_amdgcn_wmma_f32_16x16x32_bf16(false, a.v, false, b.v,
                                                          (short)0, acc, false, false);
        }
#pragma unroll
        for (int r = 0; r < 8; ++r)
            sD[(16 * wave + r + 8 * half) * 16 + nl] = acc[r];
    }
    __syncthreads();

    if (tid < 32) {
        int e = tid;
        float sc = sD[e * 16 + (e & 15)] + b5[sY[e]];
        scores[base + e] = sc;
        ebatch[base + e] = sG[e];
    }
}

// ---------------- softmax reductions (deterministic two-level online max/sum) ----------------
__device__ __forceinline__ void merge_ms(float& m, float& s, float m2, float s2) {
    if (m2 > m) { s = s * __expf(m - m2) + s2; m = m2; }
    else        { s = s + s2 * __expf(m2 - m); }
}

__global__ void reduce1(const float* __restrict__ scores,
                        float* __restrict__ pM, float* __restrict__ pS) {
    __shared__ float sm[256], ss[256];
    int tid = threadIdx.x;
    float m = -INFINITY, s = 0.f;
    for (int i = blockIdx.x * blockDim.x + tid; i < kE; i += gridDim.x * blockDim.x) {
        float x = scores[i];
        if (x > m) { s = s * __expf(m - x) + 1.f; m = x; }
        else       { s += __expf(x - m); }
    }
    sm[tid] = m; ss[tid] = s;
    __syncthreads();
    for (int off = 128; off > 0; off >>= 1) {
        if (tid < off) {
            float m2 = sm[tid + off], s2 = ss[tid + off];
            float mm = sm[tid], sc = ss[tid];
            merge_ms(mm, sc, m2, s2);
            sm[tid] = mm; ss[tid] = sc;
        }
        __syncthreads();
    }
    if (tid == 0) { pM[blockIdx.x] = sm[0]; pS[blockIdx.x] = ss[0]; }
}

__global__ void reduce2(const float* __restrict__ pM, const float* __restrict__ pS,
                        float* __restrict__ finalMS,
                        unsigned* __restrict__ segU, int* __restrict__ segA) {
    __shared__ float sm[1024], ss[1024];
    int tid = threadIdx.x;
    sm[tid] = pM[tid]; ss[tid] = pS[tid];
    __syncthreads();
    for (int off = 512; off > 0; off >>= 1) {
        if (tid < off) {
            float m2 = sm[tid + off], s2 = ss[tid + off];
            float mm = sm[tid], sc = ss[tid];
            merge_ms(mm, sc, m2, s2);
            sm[tid] = mm; ss[tid] = sc;
        }
        __syncthreads();
    }
    if (tid == 0) { finalMS[0] = sm[0]; finalMS[1] = ss[0]; }
    if (tid < kB) { segU[tid] = 0u; segA[tid] = kE; }   // init per-graph atomics
}

__global__ void probs_kernel(const float* __restrict__ scores,
                             const float* __restrict__ finalMS,
                             float* __restrict__ out) {
    int e = blockIdx.x * blockDim.x + threadIdx.x;
    if (e < kE) out[e] = __expf(scores[e] - finalMS[0]) / finalMS[1];
}

__global__ void segmax_kernel(const float* __restrict__ scores,
                              const int* __restrict__ ebatch,
                              unsigned* __restrict__ segU) {
    int e = blockIdx.x * blockDim.x + threadIdx.x;
    if (e < kE) atomicMax(&segU[ebatch[e]], enc_order(scores[e]));
}

__global__ void argmin_kernel(const float* __restrict__ scores,
                              const int* __restrict__ ebatch,
                              const unsigned* __restrict__ segU,
                              int* __restrict__ segA) {
    int e = blockIdx.x * blockDim.x + threadIdx.x;
    if (e < kE) {
        int g = ebatch[e];
        if (enc_order(scores[e]) == segU[g]) atomicMin(&segA[g], e);
    }
}

__global__ void finalize_kernel(const unsigned* __restrict__ segU,
                                const int* __restrict__ segA,
                                const float* __restrict__ finalMS,
                                float* __restrict__ out) {
    int g = blockIdx.x * blockDim.x + threadIdx.x;
    if (g < kB) {
        out[kE + g]      = __expf(dec_order(segU[g]) - finalMS[0]) / finalMS[1];
        out[kE + kB + g] = (float)segA[g];   // int action index, exact in f32
    }
}

// ---------------- launcher ----------------
extern "C" void kernel_launch(void* const* d_in, const int* in_sizes, int n_in,
                              void* d_out, int out_size, void* d_ws, size_t ws_size,
                              hipStream_t stream) {
    const float* node_reps = (const float*)d_in[0];
    const int*   edge_idx  = (const int*)d_in[1];
    const int*   batch     = (const int*)d_in[2];
    const int*   y         = (const int*)d_in[3];
    const float* W1 = (const float*)d_in[4];  const float* b1 = (const float*)d_in[5];
    const float* W2 = (const float*)d_in[6];  const float* b2 = (const float*)d_in[7];
    const float* W3 = (const float*)d_in[8];  const float* b3 = (const float*)d_in[9];
    const float* W4 = (const float*)d_in[10]; const float* b4 = (const float*)d_in[11];
    const float* W5 = (const float*)d_in[12]; const float* b5 = (const float*)d_in[13];

    char* ws = (char*)d_ws;
    unsigned short* W1T = (unsigned short*)(ws + OFF_W1T);
    unsigned short* W2T = (unsigned short*)(ws + OFF_W2T);
    unsigned short* W3T = (unsigned short*)(ws + OFF_W3T);
    unsigned short* W4T = (unsigned short*)(ws + OFF_W4T);
    unsigned short* W5T = (unsigned short*)(ws + OFF_W5T);
    float*    scores  = (float*)(ws + OFF_SCORES);
    int*      ebatch  = (int*)(ws + OFF_EBATCH);
    float*    pM      = (float*)(ws + OFF_PM);
    float*    pS      = (float*)(ws + OFF_PS);
    float*    finalMS = (float*)(ws + OFF_MS);
    unsigned* segU    = (unsigned*)(ws + OFF_SEGU);
    int*      segA    = (int*)(ws + OFF_SEGA);
    float*    out     = (float*)d_out;

    const int prepElems = 512 * 256 + 256 * 512 + 128 * 256 + 128 * 128 + 10 * 128;
    prep_weights<<<(prepElems + 255) / 256, 256, 0, stream>>>(W1, W2, W3, W4, W5,
                                                              W1T, W2T, W3T, W4T, W5T);
    edge_mlp<<<kE / 32, 256, 0, stream>>>(node_reps, edge_idx, batch, y,
                                          b1, b2, b3, b4, b5,
                                          W1T, W2T, W3T, W4T, W5T,
                                          scores, ebatch);
    reduce1<<<1024, 256, 0, stream>>>(scores, pM, pS);
    reduce2<<<1, 1024, 0, stream>>>(pM, pS, finalMS, segU, segA);
    probs_kernel<<<(kE + 255) / 256, 256, 0, stream>>>(scores, finalMS, out);
    segmax_kernel<<<(kE + 255) / 256, 256, 0, stream>>>(scores, ebatch, segU);
    argmin_kernel<<<(kE + 255) / 256, 256, 0, stream>>>(scores, ebatch, segU, segA);
    finalize_kernel<<<(kB + 255) / 256, 256, 0, stream>>>(segU, segA, finalMS, out);
}
